// attention_net_42064909697105
// MI455X (gfx1250) — matrix operand order repaired and checked by reading, not compiled, not run on hardware
//
#include <hip/hip_runtime.h>
#include <hip/hip_bf16.h>
#include <math.h>

// ---------------------------------------------------------------------------
// CDNA5 / gfx1250 wave32 WMMA implementation of the NTS-Net forward pass.
//   - all convs as implicit-GEMM with v_wmma_f32_16x16x32_bf16 (f32 accum)
//   - weights pre-converted to K-padded bf16; B tiles staged into LDS with
//     global_load_async_to_lds_b128 (ASYNCcnt) overlapping A-side im2col
//   - crop_resize fused into the part-backbone A gather (stride 32 >= k 7
//     means only 49/1024 of each crop is ever touched)
// ---------------------------------------------------------------------------

#define USE_ASYNC_LDS 1

typedef __bf16 bf16_t;
typedef __bf16 v16bf __attribute__((ext_vector_type(16)));
typedef __bf16 v8bf  __attribute__((ext_vector_type(8)));
typedef float  v8f   __attribute__((ext_vector_type(8)));

#define CATS   200
#define TOPN   4
#define IMGB   8
#define NPART  32
#define HW     448
#define PADIMG 224
#define NBOX   1614
#define OUT_RAW    0
#define OUT_CONCAT 1600
#define OUT_PART   3200
#define OUT_IDX    9600
#define OUT_PROB   9632

// ------------------------- anchor boxes (device) ---------------------------
__device__ __forceinline__ void anchor_box(int i, int& y0, int& x0, int& y1, int& x1) {
    int stride, size, ow, si_off, local;
    if (i < 1176)      { stride = 32;  size = 48;  ow = 14; si_off = 1; local = i; }
    else if (i < 1470) { stride = 64;  size = 96;  ow = 7;  si_off = 1; local = i - 1176; }
    else               { stride = 128; size = 192; ow = 4;  si_off = 0; local = i - 1470; }
    const int cells = ow * ow;
    const int blk = local / cells;
    const int pos = local - blk * cells;
    const int py = pos / ow, px = pos - py * ow;
    const float sc[3]  = {1.0f, 1.2599210498948732f, 1.5874010519681994f};
    const float arv[3] = {0.667f, 1.0f, 1.5f};
    const float s  = sc[blk / 3 + si_off];
    const float ar = arv[blk % 3];
    const float cy = py * (float)stride + 0.5f * stride;
    const float cx = px * (float)stride + 0.5f * stride;
    const float sq = sqrtf(ar);
    const float hh = size * s / sq;
    const float ww = size * s * sq;
    y0 = (int)(cy - 0.5f * hh + (float)PADIMG);
    x0 = (int)(cx - 0.5f * ww + (float)PADIMG);
    y1 = (int)(cy + 0.5f * hh + (float)PADIMG);
    x1 = (int)(cx + 0.5f * ww + (float)PADIMG);
}

__device__ __forceinline__ float xpad(const float* __restrict__ x, int b, int c, int y, int xx) {
    y -= PADIMG; xx -= PADIMG;
    if ((unsigned)y >= (unsigned)HW || (unsigned)xx >= (unsigned)HW) return 0.0f;
    return x[((b * 3 + c) * HW + y) * HW + xx];
}

// Fused crop_resize: sample pixel (c, py, px) of part image p directly from x.
__device__ __forceinline__ float sample_part(const float* __restrict__ x,
                                             const int* __restrict__ topidx,
                                             int p, int c, int py, int px) {
    const int b = p >> 2;                 // TOPN == 4
    int y0, x0, y1, x1;
    anchor_box(topidx[p], y0, x0, y1, x1);
    y1 = max(y1, y0 + 1);
    x1 = max(x1, x0 + 1);
    const float sy = (float)(y1 - 1 - y0) * (1.0f / (HW - 1));
    const float sx = (float)(x1 - 1 - x0) * (1.0f / (HW - 1));
    const float fy = (float)y0 + py * sy;
    const float fx = (float)x0 + px * sx;
    const int yl = (int)floorf(fy);
    const int yh = min(yl + 1, y1 - 1);
    const float wy = fy - (float)yl;
    const int xl = (int)floorf(fx);
    const int xh = min(xl + 1, x1 - 1);
    const float wx = fx - (float)xl;
    const float vll = xpad(x, b, c, yl, xl), vlh = xpad(x, b, c, yl, xh);
    const float vhl = xpad(x, b, c, yh, xl), vhh = xpad(x, b, c, yh, xh);
    return (1.0f - wy) * ((1.0f - wx) * vll + wx * vlh)
         +          wy * ((1.0f - wx) * vhl + wx * vhh);
}

// ----------------- f32 -> bf16 weight conversion (K padded to 32) ----------
__global__ void wconv_kernel(const float* __restrict__ w, bf16_t* __restrict__ out,
                             int Cout, int Ktot, int Kpad) {
    const int i = blockIdx.x * blockDim.x + threadIdx.x;
    if (i >= Cout * Kpad) return;
    const int n = i / Kpad, k = i - n * Kpad;
    out[i] = (bf16_t)((k < Ktot) ? w[(size_t)n * Ktot + k] : 0.0f);
}

// ------------------------- implicit-GEMM WMMA conv -------------------------
struct ConvArgs {
    const float*  in;      // NCHW input (or original x in crop mode)
    const bf16_t* wbf;     // [Cout][Kpad] bf16, zero-padded K
    const float*  bias;    // [Cout]
    float*        out;     // [Nimg][Cout][OH][OW]
    const int*    topidx;  // crop mode: 32 picked anchor indices
    int Nimg, Cin, Hin, Win, Cout, KH, KW, stride, pad, OH, OW, Ktot, Kpad, M, relu, crop;
};

#define LDA 40   // 32 bf16 + 8 pad; 80B row stride keeps 16B alignment

// Incremental im2col gather of 8 consecutive K values (one div per run).
__device__ __forceinline__ void im2col8(const ConvArgs& p, int img, int ybase, int xbase,
                                        int k0, float* vals) {
    const int KHW = p.KH * p.KW;
    int c  = k0 / KHW;
    int rr = k0 - c * KHW;
    int ky = rr / p.KW;
    int kx = rr - ky * p.KW;
    #pragma unroll
    for (int i = 0; i < 8; ++i) {
        float v = 0.0f;
        if (c < p.Cin) {                       // c >= Cin  <=>  k >= Ktot
            const int iy = ybase + ky;
            const int ix = xbase + kx;
            if ((unsigned)iy < (unsigned)p.Hin && (unsigned)ix < (unsigned)p.Win) {
                v = p.crop ? sample_part(p.in, p.topidx, img, c, iy, ix)
                           : p.in[((img * p.Cin + c) * p.Hin + iy) * p.Win + ix];
            }
        }
        vals[i] = v;
        // carry-chain advance kx -> ky -> c
        if (++kx == p.KW) { kx = 0; if (++ky == p.KH) { ky = 0; ++c; } }
    }
}

__launch_bounds__(256)
__global__ void conv_wmma_kernel(ConvArgs p) {
    __shared__ __align__(16) bf16_t As[128 * LDA];  // A tile : 128 rows x 32 K
    __shared__ __align__(16) bf16_t Bs[128 * LDA];  // Bt tile: 128 cols x 32 K

    const int tid   = threadIdx.x;
    const int lane  = tid & 31;
    const int wid   = tid >> 5;          // 8 waves: 4 (M) x 2 (N)
    const int wm    = wid >> 1;          // 0..3  -> 32 rows each
    const int wn    = wid & 1;           // 0..1  -> 64 cols each
    const int lrow  = lane & 15;
    const int lhalf = lane >> 4;
    const int blockM = blockIdx.x * 128;
    const int blockN = blockIdx.y * 128;
    const int OHOW = p.OH * p.OW;

    v8f acc[2][4];
    #pragma unroll
    for (int sm = 0; sm < 2; ++sm)
        #pragma unroll
        for (int sn = 0; sn < 4; ++sn) acc[sm][sn] = (v8f){};

    // A staging role: 2 threads per row, 16 consecutive K each (two runs of 8)
    const int arow  = tid >> 1;
    const int akoff = (tid & 1) << 4;
    const int am    = blockM + arow;
    const bool amok = (am < p.M);
    int aimg = 0, ybase = 0, xbase = 0;
    if (amok) {
        aimg = am / OHOW;
        const int pos = am - aimg * OHOW;
        const int oy = pos / p.OW;
        const int ox = pos - oy * p.OW;
        ybase = oy * p.stride - p.pad;
        xbase = ox * p.stride - p.pad;
    }

#if USE_ASYNC_LDS
    const unsigned bs_base = (unsigned)(uintptr_t)(&Bs[0]);   // LDS byte offset
#endif

    const int ksteps = p.Kpad >> 5;
    for (int kt = 0; kt < ksteps; ++kt) {
        const int kb = kt << 5;

        // ---- stage Bt: 128 rows x 64B of bf16 weights, global -> LDS ----
#if USE_ASYNC_LDS
        // async DMA into LDS (ASYNCcnt), overlapped with A-side im2col below
        #pragma unroll
        for (int s = tid; s < 512; s += 256) {
            const int n   = s >> 2;
            const int seg = s & 3;
            const unsigned g_off = (unsigned)(((size_t)(blockN + n) * p.Kpad + kb) * 2 + seg * 16);
            const unsigned l_off = bs_base + (unsigned)(n * LDA * 2 + seg * 16);
            asm volatile("global_load_async_to_lds_b128 %0, %1, %2"
                         :: "v"(l_off), "v"(g_off), "s"(p.wbf) : "memory");
        }
#else
        #pragma unroll
        for (int s = tid; s < 512; s += 256) {
            const int n   = s >> 2;
            const int seg = s & 3;
            const v8bf* gp = (const v8bf*)(p.wbf + (size_t)(blockN + n) * p.Kpad + kb + seg * 8);
            *(v8bf*)&Bs[n * LDA + seg * 8] = *gp;
        }
#endif

        // ---- stage A: im2col gather (optionally fused bilinear crop) ----
        {
            float v0[8], v1[8];
            if (amok) {
                im2col8(p, aimg, ybase, xbase, kb + akoff,     v0);
                im2col8(p, aimg, ybase, xbase, kb + akoff + 8, v1);
            } else {
                #pragma unroll
                for (int i = 0; i < 8; ++i) { v0[i] = 0.0f; v1[i] = 0.0f; }
            }
            #pragma unroll
            for (int i = 0; i < 8; ++i) As[arow * LDA + akoff + i]     = (bf16_t)v0[i];
            #pragma unroll
            for (int i = 0; i < 8; ++i) As[arow * LDA + akoff + 8 + i] = (bf16_t)v1[i];
        }

#if USE_ASYNC_LDS
        asm volatile("s_wait_asynccnt 0" ::: "memory");
#endif
        __syncthreads();

        // ---- fragments per ISA 16-bit layouts (two 16B ds loads each) ----
        union frag { v16bf v; v8bf h[2]; };
        frag a[2], b[4];
        #pragma unroll
        for (int sm = 0; sm < 2; ++sm) {
            const bf16_t* rp = &As[(wm * 32 + sm * 16 + lrow) * LDA];
            // A: lanes 0-15 -> K {0..7,16..23}; lanes 16-31 -> K {8..15,24..31}
            a[sm].h[0] = *(const v8bf*)(rp + lhalf * 8);
            a[sm].h[1] = *(const v8bf*)(rp + 16 + lhalf * 8);
        }
        #pragma unroll
        for (int sn = 0; sn < 4; ++sn) {
            const bf16_t* rp = &Bs[(wn * 64 + sn * 16 + lrow) * LDA];
            // B: lanes 0-15 -> K {0..7,16..23}; lanes 16-31 -> K {8..15,24..31}
            b[sn].h[0] = *(const v8bf*)(rp + lhalf * 8);
            b[sn].h[1] = *(const v8bf*)(rp + 16 + lhalf * 8);
        }
        #pragma unroll
        for (int sm = 0; sm < 2; ++sm)
            #pragma unroll
            for (int sn = 0; sn < 4; ++sn)
                acc[sm][sn] = __builtin_amdgcn_wmma_f32_16x16x32_bf16(
                    false, a[sm].v, false, b[sn].v, (short)0, acc[sm][sn], false, false);
        __syncthreads();
    }

    // ---- epilogue: bias + relu, NCHW scatter. D: VGPR e -> M=e+8*lhalf, N=lrow
    #pragma unroll
    for (int sm = 0; sm < 2; ++sm) {
        #pragma unroll
        for (int sn = 0; sn < 4; ++sn) {
            const int n = blockN + wn * 64 + sn * 16 + lrow;
            if (n < p.Cout) {
                const float bv = p.bias[n];
                #pragma unroll
                for (int e = 0; e < 8; ++e) {
                    const int m = blockM + wm * 32 + sm * 16 + lhalf * 8 + e;
                    if (m < p.M) {
                        float v = acc[sm][sn][e] + bv;
                        if (p.relu) v = fmaxf(v, 0.0f);
                        const int img = m / OHOW;
                        const int pos = m - img * OHOW;
                        p.out[((size_t)img * p.Cout + n) * OHOW + pos] = v;
                    }
                }
            }
        }
    }
}

// ---------------------------- spatial mean ---------------------------------
__global__ void mean196_kernel(const float* __restrict__ f, float* __restrict__ g, int count) {
    const int i = blockIdx.x * blockDim.x + threadIdx.x;
    if (i >= count) return;
    const float* p = f + (size_t)i * 196;
    float s = 0.0f;
    #pragma unroll 4
    for (int j = 0; j < 196; ++j) s += p[j];
    g[i] = s * (1.0f / 196.0f);
}

// -------------------- tidy 1x1 convs -> rpn_score --------------------------
__global__ void tidy_kernel(const float* __restrict__ d1, const float* __restrict__ d2,
                            const float* __restrict__ d3,
                            const float* __restrict__ w1, const float* __restrict__ b1,
                            const float* __restrict__ w2, const float* __restrict__ b2,
                            const float* __restrict__ w3, const float* __restrict__ b3,
                            float* __restrict__ rpn) {
    const int i = blockIdx.x * blockDim.x + threadIdx.x;
    if (i >= IMGB * NBOX) return;
    const int b = i / NBOX;
    const int j = i - b * NBOX;
    float s;
    if (j < 1176) {
        const int c = j / 196, pos = j - c * 196;
        const float* in = d1 + (size_t)b * 128 * 196;
        s = b1[c];
        for (int ci = 0; ci < 128; ++ci) s += in[ci * 196 + pos] * w1[c * 128 + ci];
    } else if (j < 1470) {
        const int jj = j - 1176;
        const int c = jj / 49, pos = jj - c * 49;
        const float* in = d2 + (size_t)b * 128 * 49;
        s = b2[c];
        for (int ci = 0; ci < 128; ++ci) s += in[ci * 49 + pos] * w2[c * 128 + ci];
    } else {
        const int jj = j - 1470;
        const int c = jj / 16, pos = jj - c * 16;
        const float* in = d3 + (size_t)b * 128 * 16;
        s = b3[c];
        for (int ci = 0; ci < 128; ++ci) s += in[ci * 16 + pos] * w3[c * 128 + ci];
    }
    rpn[i] = s;
}

// ------------------------------- NMS ---------------------------------------
__launch_bounds__(256)
__global__ void nms_kernel(const float* __restrict__ rpn,
                           float* __restrict__ out_idx, float* __restrict__ out_prob,
                           int* __restrict__ topidx) {
    const int b = blockIdx.x;
    const int tid = threadIdx.x;
    __shared__ float sc[NBOX];
    __shared__ float red_v[256];
    __shared__ int   red_i[256];
    for (int i = tid; i < NBOX; i += 256) sc[i] = rpn[b * NBOX + i];
    __syncthreads();

    int picks[TOPN];
    #pragma unroll
    for (int t = 0; t < TOPN; ++t) {
        float best = -INFINITY; int bi = 0x7fffffff;
        for (int i = tid; i < NBOX; i += 256) {
            const float v = sc[i];
            if (v > best || (v == best && i < bi)) { best = v; bi = i; }
        }
        red_v[tid] = best; red_i[tid] = bi;
        __syncthreads();
        for (int s = 128; s > 0; s >>= 1) {
            if (tid < s) {
                const float v2 = red_v[tid + s]; const int i2 = red_i[tid + s];
                if (v2 > red_v[tid] || (v2 == red_v[tid] && i2 < red_i[tid])) {
                    red_v[tid] = v2; red_i[tid] = i2;
                }
            }
            __syncthreads();
        }
        const int pick = red_i[0];
        picks[t] = pick;
        __syncthreads();

        int py0, px0, py1, px1;
        anchor_box(pick, py0, px0, py1, px1);
        const float fy0 = (float)py0, fx0 = (float)px0, fy1 = (float)py1, fx1 = (float)px1;
        const float area_p = (fy1 - fy0) * (fx1 - fx0);
        for (int i = tid; i < NBOX; i += 256) {
            int jy0, jx0, jy1, jx1;
            anchor_box(i, jy0, jx0, jy1, jx1);
            const float gy0 = (float)jy0, gx0 = (float)jx0, gy1 = (float)jy1, gx1 = (float)jx1;
            const float ty = fmaxf(gy0, fy0), tx = fmaxf(gx0, fx0);
            const float by = fminf(gy1, fy1), bx = fminf(gx1, fx1);
            const float wy = by - ty, wx = bx - tx;
            const float inter = (wy < 0.0f || wx < 0.0f) ? 0.0f : wy * wx;
            const float area_j = (gy1 - gy0) * (gx1 - gx0);
            const float iou = inter / (area_j + area_p - inter);
            if (iou >= 0.25f) sc[i] = -INFINITY;
        }
        __syncthreads();
    }

    if (tid < TOPN) {
        const int pk = picks[tid];
        out_idx[b * TOPN + tid]  = (float)pk;
        out_prob[b * TOPN + tid] = rpn[b * NBOX + pk];
        topidx[b * TOPN + tid]   = pk;
    }
}

// ----------------- wave-per-output FC (wave32 shuffle reduce) ---------------
__global__ void fc_kernel(const float* __restrict__ A, const float* __restrict__ W,
                          const float* __restrict__ bias, float* __restrict__ out,
                          int M, int N, int K) {
    const int gwave = (blockIdx.x * blockDim.x + threadIdx.x) >> 5;
    const int lane  = threadIdx.x & 31;
    if (gwave >= M * N) return;
    const int m = gwave / N, n = gwave - m * N;
    const float* a = A + (size_t)m * K;
    const float* w = W + (size_t)n * K;
    float s = 0.0f;
    for (int k = lane; k < K; k += 32) s += a[k] * w[k];
    #pragma unroll
    for (int off = 16; off > 0; off >>= 1) s += __shfl_down(s, off, 32);
    if (lane == 0) out[(size_t)m * N + n] = s + bias[n];
}

__global__ void fc2_kernel(const float* __restrict__ A1, int K1,
                           const float* __restrict__ A2, int K2,
                           const float* __restrict__ W, const float* __restrict__ bias,
                           float* __restrict__ out, int M, int N) {
    const int gwave = (blockIdx.x * blockDim.x + threadIdx.x) >> 5;
    const int lane  = threadIdx.x & 31;
    if (gwave >= M * N) return;
    const int m = gwave / N, n = gwave - m * N;
    const int K = K1 + K2;
    const float* w = W + (size_t)n * K;
    float s = 0.0f;
    for (int k = lane; k < K; k += 32) {
        const float av = (k < K1) ? A1[(size_t)m * K1 + k] : A2[(size_t)m * K2 + (k - K1)];
        s += av * w[k];
    }
    #pragma unroll
    for (int off = 16; off > 0; off >>= 1) s += __shfl_down(s, off, 32);
    if (lane == 0) out[(size_t)m * N + n] = s + bias[n];
}

// ------------------------------ host side ----------------------------------
static void launch_conv(const ConvArgs& a, hipStream_t s) {
    dim3 grid((a.M + 127) / 128, (a.Cout + 127) / 128);
    conv_wmma_kernel<<<grid, 256, 0, s>>>(a);
}

extern "C" void kernel_launch(void* const* d_in, const int* in_sizes, int n_in,
                              void* d_out, int out_size, void* d_ws, size_t ws_size,
                              hipStream_t stream) {
    const float* x        = (const float*)d_in[0];
    const float* stem_w   = (const float*)d_in[1];
    const float* stem_b   = (const float*)d_in[2];
    const float* fc_w     = (const float*)d_in[3];
    const float* fc_b     = (const float*)d_in[4];
    const float* down1_w  = (const float*)d_in[5];
    const float* down1_b  = (const float*)d_in[6];
    const float* down2_w  = (const float*)d_in[7];
    const float* down2_b  = (const float*)d_in[8];
    const float* down3_w  = (const float*)d_in[9];
    const float* down3_b  = (const float*)d_in[10];
    const float* tidy1_w  = (const float*)d_in[11];
    const float* tidy1_b  = (const float*)d_in[12];
    const float* tidy2_w  = (const float*)d_in[13];
    const float* tidy2_b  = (const float*)d_in[14];
    const float* tidy3_w  = (const float*)d_in[15];
    const float* tidy3_b  = (const float*)d_in[16];
    const float* concat_w = (const float*)d_in[17];
    const float* concat_b = (const float*)d_in[18];
    const float* part_w   = (const float*)d_in[19];
    const float* part_b   = (const float*)d_in[20];
    float* out = (float*)d_out;
    float* ws  = (float*)d_ws;

    // ---- workspace layout (f32 region, then bf16 weight images) ----
    float* fbuf   = ws;                                  // 32*2048*196 (f8 then f_part)
    float* g8     = fbuf   + (size_t)NPART * 2048 * 196;
    float* gpart  = g8     + (size_t)IMGB * 2048;
    float* d1     = gpart  + (size_t)NPART * 2048;
    float* d2     = d1     + (size_t)IMGB * 128 * 196;
    float* d3     = d2     + (size_t)IMGB * 128 * 49;
    float* rpn    = d3     + (size_t)IMGB * 128 * 16;
    int*   topidx = (int*)(rpn + (size_t)IMGB * NBOX);
    bf16_t* stem_bf  = (bf16_t*)(topidx + 32);           // 2048 x 160
    bf16_t* down1_bf = stem_bf  + (size_t)2048 * 160;    // 128 x 18432
    bf16_t* down2_bf = down1_bf + (size_t)128 * 18432;   // 128 x 1152
    bf16_t* down3_bf = down2_bf + (size_t)128 * 1152;    // 128 x 1152

    // ---- one-shot weight conversions (f32 -> bf16, K padded to 32) ----
    wconv_kernel<<<(2048 * 160 + 255) / 256, 256, 0, stream>>>(stem_w, stem_bf, 2048, 147, 160);
    wconv_kernel<<<(128 * 18432 + 255) / 256, 256, 0, stream>>>(down1_w, down1_bf, 128, 18432, 18432);
    wconv_kernel<<<(128 * 1152 + 255) / 256, 256, 0, stream>>>(down2_w, down2_bf, 128, 1152, 1152);
    wconv_kernel<<<(128 * 1152 + 255) / 256, 256, 0, stream>>>(down3_w, down3_bf, 128, 1152, 1152);

    // 1) stem conv on x -> f8 (relu)
    ConvArgs c{};
    c.in = x; c.wbf = stem_bf; c.bias = stem_b; c.out = fbuf; c.topidx = nullptr;
    c.Nimg = IMGB; c.Cin = 3; c.Hin = HW; c.Win = HW; c.Cout = 2048;
    c.KH = 7; c.KW = 7; c.stride = 32; c.pad = 3; c.OH = 14; c.OW = 14;
    c.Ktot = 147; c.Kpad = 160; c.M = IMGB * 196; c.relu = 1; c.crop = 0;
    launch_conv(c, stream);

    // 2) global_feat
    mean196_kernel<<<(IMGB * 2048 + 255) / 256, 256, 0, stream>>>(fbuf, g8, IMGB * 2048);

    // 3) raw_logits
    fc_kernel<<<(IMGB * CATS * 32 + 255) / 256, 256, 0, stream>>>(
        g8, fc_w, fc_b, out + OUT_RAW, IMGB, CATS, 2048);

    // 4) down1 (K = 2048*9 = 18432)
    c.in = fbuf; c.wbf = down1_bf; c.bias = down1_b; c.out = d1;
    c.Cin = 2048; c.Hin = 14; c.Win = 14; c.Cout = 128;
    c.KH = 3; c.KW = 3; c.stride = 1; c.pad = 1; c.OH = 14; c.OW = 14;
    c.Ktot = 18432; c.Kpad = 18432; c.M = IMGB * 196; c.relu = 1; c.crop = 0;
    launch_conv(c, stream);

    // 5) down2
    c.in = d1; c.wbf = down2_bf; c.bias = down2_b; c.out = d2;
    c.Cin = 128; c.Hin = 14; c.Win = 14; c.stride = 2; c.OH = 7; c.OW = 7;
    c.Ktot = 1152; c.Kpad = 1152; c.M = IMGB * 49;
    launch_conv(c, stream);

    // 6) down3
    c.in = d2; c.wbf = down3_bf; c.bias = down3_b; c.out = d3;
    c.Hin = 7; c.Win = 7; c.OH = 4; c.OW = 4; c.M = IMGB * 16;
    launch_conv(c, stream);

    // 7) tidy -> rpn_score
    tidy_kernel<<<(IMGB * NBOX + 255) / 256, 256, 0, stream>>>(
        d1, d2, d3, tidy1_w, tidy1_b, tidy2_w, tidy2_b, tidy3_w, tidy3_b, rpn);

    // 8) NMS -> top_n_idx / top_n_prob
    nms_kernel<<<IMGB, 256, 0, stream>>>(rpn, out + OUT_IDX, out + OUT_PROB, topidx);

    // 9) stem conv on fused bilinear crops -> f_part (reuses fbuf)
    c.in = x; c.wbf = stem_bf; c.bias = stem_b; c.out = fbuf; c.topidx = topidx;
    c.Nimg = NPART; c.Cin = 3; c.Hin = HW; c.Win = HW; c.Cout = 2048;
    c.KH = 7; c.KW = 7; c.stride = 32; c.pad = 3; c.OH = 14; c.OW = 14;
    c.Ktot = 147; c.Kpad = 160; c.M = NPART * 196; c.relu = 1; c.crop = 1;
    launch_conv(c, stream);

    // 10) part_feat
    mean196_kernel<<<(NPART * 2048 + 255) / 256, 256, 0, stream>>>(fbuf, gpart, NPART * 2048);

    // 11) concat_logits
    fc2_kernel<<<(IMGB * CATS * 32 + 255) / 256, 256, 0, stream>>>(
        gpart, TOPN * 2048, g8, 2048, concat_w, concat_b, out + OUT_CONCAT, IMGB, CATS);

    // 12) part_logits
    fc_kernel<<<(NPART * CATS * 32 + 255) / 256, 256, 0, stream>>>(
        gpart, part_w, part_b, out + OUT_PART, NPART, CATS, 2048);
}
